// SelfMultiheadAttention_35107062677954
// MI455X (gfx1250) — compile-verified, matrix operand order checked by reading, not run on hardware
//
#include <hip/hip_runtime.h>
#include <hip/hip_bf16.h>

// ---------------------------------------------------------------------------
// SelfMultiheadAttention for MI455X (gfx1250), wave32 + WMMA bf16 + TDM.
// N_NODE=512, N_GRAPH=16, EMBED=512, HEADS=32, HEAD_DIM=16, SCALING=0.25
// ---------------------------------------------------------------------------

typedef __attribute__((ext_vector_type(16))) __bf16 v16bf;
typedef __attribute__((ext_vector_type(8)))  float  v8f;
typedef __attribute__((ext_vector_type(4)))  unsigned int v4u;
typedef __attribute__((ext_vector_type(8)))  int    v8i;
typedef __attribute__((ext_vector_type(4)))  int    v4i;

#define N_NODE   512
#define N_GRAPH  16
#define EMBED    512
#define NHEADS   32
#define HDIM     16
#define NROWS    (N_NODE * N_GRAPH)   // 8192 flattened (n,b) rows
#define GTOT     (N_GRAPH * NHEADS)   // 512 batched heads

// LDS layout of attn_kernel (dynamic LDS, base offset 0):
//   Ks [512][16] bf16 : 16384 B
//   Vs [512][16] bf16 : 16384 B
//   Qs [128][16] bf16 :  4096 B
//   Ss [128][512] f32 : 262144 B   -> byte offset 36864
#define LDS_SS_OFF ((2 * N_NODE * HDIM + 128 * HDIM) * 2)

__device__ __forceinline__ unsigned short f32_to_bf16(float f) {
  unsigned int u = __builtin_bit_cast(unsigned int, f);
  unsigned int r = u + 0x7FFFu + ((u >> 16) & 1u);   // round-to-nearest-even
  return (unsigned short)(r >> 16);
}
__device__ __forceinline__ __bf16 bf16_bits(unsigned short b) {
  return __builtin_bit_cast(__bf16, b);
}

// 16-bit A-matrix 16x32 per-lane K index (ISA 7.12.2):
// VGPR 0-3: lanes0-15 K=2v..2v+1, lanes16-31 K=8+2v..; VGPR 4-7: +16.
__device__ __forceinline__ int a_frag_k(int e, int lane) {
  int v = e >> 1, p = e & 1;
  int kb = (lane & 16) ? 8 : 0;
  return (v < 4) ? (kb + 2 * v + p) : (16 + kb + 2 * (v - 4) + p);
}

__device__ __forceinline__ v8f wmma_bf16(v16bf a, v16bf b, v8f c) {
  return __builtin_amdgcn_wmma_f32_16x16x32_bf16(false, a, false, b,
                                                 (short)0, c, false, false);
}
__device__ __forceinline__ v8f zero8() {
  v8f z;
#pragma unroll
  for (int i = 0; i < 8; ++i) z[i] = 0.0f;
  return z;
}

// ---------------------------------------------------------------------------
// Kernel 1: fused QKV projection.
//   qkv[r][f] = sum_e query[r][e] * in_proj_w[f][e] + in_proj_b[f]
// r = n*16+b (query is (n,b,e) row-major -> rows are contiguous), f in [0,1536).
// Scatter to q/k/v workspaces in (g, n, d) bf16 layout, q scaled by 0.25.
// Block tile 128x64, 256 threads = 8 waves; each wave: 16 rows x 64 cols.
// ---------------------------------------------------------------------------
__global__ __launch_bounds__(256)
void qkv_gemm_kernel(const float* __restrict__ query,
                     const float* __restrict__ W,
                     const float* __restrict__ bias,
                     unsigned short* __restrict__ qo,
                     unsigned short* __restrict__ ko,
                     unsigned short* __restrict__ vo) {
  __shared__ unsigned short As[128 * 32];   // [row][k]
  __shared__ unsigned short Bs[64 * 32];    // [n][k]
  const int tid  = threadIdx.x;
  const int lane = tid & 31, wave = tid >> 5;
  const int Mbase = blockIdx.x * 128;
  const int Nbase = blockIdx.y * 64;

  v8f acc[4];
#pragma unroll
  for (int t = 0; t < 4; ++t) acc[t] = zero8();

  const int m16 = lane & 15;
  for (int kk = 0; kk < EMBED; kk += 32) {
    __syncthreads();
#pragma unroll
    for (int idx = tid; idx < 128 * 32; idx += 256) {
      int r = idx >> 5, k = idx & 31;
      As[idx] = f32_to_bf16(query[(size_t)(Mbase + r) * EMBED + kk + k]);
    }
#pragma unroll
    for (int idx = tid; idx < 64 * 32; idx += 256) {
      int n = idx >> 5, k = idx & 31;
      Bs[idx] = f32_to_bf16(W[(size_t)(Nbase + n) * EMBED + kk + k]);
    }
    __syncthreads();

    v16bf a;
    const unsigned short* arow = &As[(wave * 16 + m16) * 32];
#pragma unroll
    for (int e = 0; e < 16; ++e) a[e] = bf16_bits(arow[a_frag_k(e, lane)]);
#pragma unroll
    for (int t = 0; t < 4; ++t) {
      v16bf b;
      const unsigned short* brow = &Bs[(t * 16 + m16) * 32];
#pragma unroll
      for (int e = 0; e < 16; ++e) b[e] = bf16_bits(brow[a_frag_k(e, lane)]);
      acc[t] = wmma_bf16(a, b, acc[t]);
    }
  }

  const int moff = (lane & 16) ? 8 : 0;
#pragma unroll
  for (int t = 0; t < 4; ++t) {
    const int f  = Nbase + t * 16 + m16;
    const float bv = bias[f];
    const int fl = f & 511;
    const int h = fl >> 4, d = fl & 15;
#pragma unroll
    for (int i = 0; i < 8; ++i) {
      const int R  = Mbase + wave * 16 + i + moff;
      const int nn = R >> 4, bb = R & 15;
      const int g  = bb * NHEADS + h;
      const size_t o = ((size_t)g * N_NODE + nn) * HDIM + d;
      const float val = acc[t][i] + bv;
      if (f < 512)        qo[o] = f32_to_bf16(val * 0.25f);  // SCALING
      else if (f < 1024)  ko[o] = f32_to_bf16(val);
      else                vo[o] = f32_to_bf16(val);
    }
  }
}

// ---------------------------------------------------------------------------
// Kernel 2: attention for one (g, 128-row strip).
// 256 threads = 8 waves; wave w owns rows [w*16, w*16+16) of the strip.
// Dynamic LDS: K_g 16KB + V_g 16KB + Q strip 4KB + S strip 256KB = 292KB.
// The 512MB attn_bias stream is pulled HBM->LDS by the Tensor Data Mover
// (one 128x512 fp32 tile per block), then consumed as WMMA C fragments.
// ---------------------------------------------------------------------------
__global__ __launch_bounds__(256)
void attn_kernel(const unsigned short* __restrict__ qw,
                 const unsigned short* __restrict__ kw,
                 const unsigned short* __restrict__ vw,
                 const float* __restrict__ attn_bias,
                 unsigned short* __restrict__ attn_out) {
  extern __shared__ unsigned char smem[];
  unsigned short* Ks = (unsigned short*)smem;           // [512][16]
  unsigned short* Vs = Ks + N_NODE * HDIM;              // [512][16]
  unsigned short* Qs = Vs + N_NODE * HDIM;              // [128][16]
  float*          Ss = (float*)(Qs + 128 * HDIM);       // [128][512]

  const int g = blockIdx.x;
  const int rowbase = blockIdx.y * 128;
  const int tid = threadIdx.x, lane = tid & 31, wave = tid >> 5;
  const size_t gbase = (size_t)g * N_NODE * HDIM;
  const float* bias_tile =
      attn_bias + ((size_t)g * N_NODE + rowbase) * N_NODE;

#if __has_builtin(__builtin_amdgcn_tensor_load_to_lds) && \
    __has_builtin(__builtin_amdgcn_s_wait_tensorcnt)
  // --- TDM: async DMA of the 128x512 fp32 bias tile into Ss -----------------
  if (wave == 0) {
    const unsigned long long ga = (unsigned long long)(const void*)bias_tile;
    v4u g0;
    g0[0] = 1u;                                   // count=1, user descriptor
    g0[1] = (unsigned)LDS_SS_OFF;                 // lds_addr (bytes)
    g0[2] = (unsigned)(ga & 0xFFFFFFFFu);         // global_addr[31:0]
    g0[3] = (unsigned)((ga >> 32) & 0x1FFFFFFu)   // global_addr[56:32]
            | (2u << 30);                         // type = 2 ("image")
    v8i g1;
    g1[0] = 0x20000;                              // data_size=2 (4B), no mask
    g1[1] = (N_NODE << 16);                       // tensor_dim0 = 512
    g1[2] = (128 << 16);                          // tensor_dim1 = 128 rows
    g1[3] = (N_NODE << 16);                       // tile_dim0 = 512
    g1[4] = 128;                                  // tile_dim1 = 128
    g1[5] = N_NODE;                               // tensor_dim0_stride = 512
    g1[6] = 0;
    g1[7] = 0;
    v4i z4 = {0, 0, 0, 0};
#if __clang_major__ >= 23
    v8i z8 = {0, 0, 0, 0, 0, 0, 0, 0};
    __builtin_amdgcn_tensor_load_to_lds(g0, g1, z4, z4, z8, 0);
#else
    __builtin_amdgcn_tensor_load_to_lds(g0, g1, z4, z4, 0);
#endif
  }
#endif

  // --- stage K, V, Q (bf16) while the TDM streams the bias tile -------------
#pragma unroll
  for (int idx = tid; idx < N_NODE * HDIM; idx += 256) {
    Ks[idx] = kw[gbase + idx];
    Vs[idx] = vw[gbase + idx];
  }
#pragma unroll
  for (int idx = tid; idx < 128 * HDIM; idx += 256)
    Qs[idx] = qw[gbase + (size_t)rowbase * HDIM + idx];

#if __has_builtin(__builtin_amdgcn_tensor_load_to_lds) && \
    __has_builtin(__builtin_amdgcn_s_wait_tensorcnt)
  if (wave == 0) __builtin_amdgcn_s_wait_tensorcnt(0);
#else
  // Fallback: coalesced cooperative copy of the bias tile (float4 wide).
  {
    const float4* src = (const float4*)bias_tile;
    float4* dst = (float4*)Ss;
    for (int idx = tid; idx < 128 * N_NODE / 4; idx += 256) dst[idx] = src[idx];
  }
#endif
  __syncthreads();

  const int m16  = lane & 15;
  const int moff = (lane & 16) ? 8 : 0;
  float* Srow = Ss + (size_t)wave * 16 * N_NODE;

  // --- scores: S = Q*K^T + bias (head dim 16, K zero-padded to 32) ---------
  v16bf aq;
#pragma unroll
  for (int e = 0; e < 16; ++e) {
    int k = a_frag_k(e, lane);
    aq[e] = (k < HDIM) ? bf16_bits(Qs[(wave * 16 + m16) * HDIM + k])
                       : bf16_bits((unsigned short)0);
  }
  for (int j = 0; j < 32; ++j) {            // 32 column tiles of 16 key-nodes
    v16bf bk;
#pragma unroll
    for (int e = 0; e < 16; ++e) {
      int k = a_frag_k(e, lane);
      bk[e] = (k < HDIM) ? bf16_bits(Ks[(j * 16 + m16) * HDIM + k])
                         : bf16_bits((unsigned short)0);
    }
    v8f cb;
#pragma unroll
    for (int i = 0; i < 8; ++i)             // bias tile (LDS) -> C fragment
      cb[i] = Srow[(i + moff) * N_NODE + j * 16 + m16];
    cb = wmma_bf16(aq, bk, cb);
#pragma unroll
    for (int i = 0; i < 8; ++i)
      Srow[(i + moff) * N_NODE + j * 16 + m16] = cb[i];
  }
  __syncthreads();

  // --- softmax: each wave softmaxes its own 16 rows ------------------------
  for (int r = 0; r < 16; ++r) {
    float* row = Srow + r * N_NODE;
    float mx = -1e30f;
    for (int c = lane; c < N_NODE; c += 32) mx = fmaxf(mx, row[c]);
#pragma unroll
    for (int off = 16; off; off >>= 1) mx = fmaxf(mx, __shfl_xor(mx, off, 32));
    float sum = 0.0f;
    for (int c = lane; c < N_NODE; c += 32) {
      float e = __expf(row[c] - mx);
      row[c] = e;
      sum += e;
    }
#pragma unroll
    for (int off = 16; off; off >>= 1) sum += __shfl_xor(sum, off, 32);
    const float inv = 1.0f / sum;
    for (int c = lane; c < N_NODE; c += 32) row[c] *= inv;
  }
  __syncthreads();

  // --- out = P (16x512) @ V (512x16), K chunks of 32 nodes -----------------
  v8f o = zero8();
  for (int j = 0; j < 16; ++j) {
    v16bf ap, bv;
#pragma unroll
    for (int e = 0; e < 16; ++e) {
      int k = a_frag_k(e, lane);
      ap[e] = bf16_bits(f32_to_bf16(Srow[m16 * N_NODE + j * 32 + k]));
      bv[e] = bf16_bits(Vs[(j * 32 + k) * HDIM + m16]);
    }
    o = wmma_bf16(ap, bv, o);
  }

  // store attn in (n, b, h*16+d) bf16 row-major (8192 x 512) for out-proj
  const int bb = g >> 5, h = g & 31;
#pragma unroll
  for (int i = 0; i < 8; ++i) {
    const int nrow = rowbase + wave * 16 + i + moff;
    attn_out[((size_t)nrow * N_GRAPH + bb) * EMBED + h * HDIM + m16] =
        f32_to_bf16(o[i]);
  }
}

// ---------------------------------------------------------------------------
// Kernel 3: output projection + bias + node mask. Same tiling as kernel 1.
// out[r][f] = mask[b][n] ? sum_e attn[r][e]*out_w[f][e] + out_b[f] : 0
// ---------------------------------------------------------------------------
__global__ __launch_bounds__(256)
void out_proj_kernel(const unsigned short* __restrict__ A,
                     const float* __restrict__ W,
                     const float* __restrict__ bias,
                     const unsigned char* __restrict__ mask,
                     float* __restrict__ out) {
  __shared__ unsigned short As[128 * 32];
  __shared__ unsigned short Bs[64 * 32];
  const int tid  = threadIdx.x;
  const int lane = tid & 31, wave = tid >> 5;
  const int Mbase = blockIdx.x * 128;
  const int Nbase = blockIdx.y * 64;

  v8f acc[4];
#pragma unroll
  for (int t = 0; t < 4; ++t) acc[t] = zero8();

  const int m16 = lane & 15;
  for (int kk = 0; kk < EMBED; kk += 32) {
    __syncthreads();
#pragma unroll
    for (int idx = tid; idx < 128 * 32; idx += 256) {
      int r = idx >> 5, k = idx & 31;
      As[idx] = A[(size_t)(Mbase + r) * EMBED + kk + k];
    }
#pragma unroll
    for (int idx = tid; idx < 64 * 32; idx += 256) {
      int n = idx >> 5, k = idx & 31;
      Bs[idx] = f32_to_bf16(W[(size_t)(Nbase + n) * EMBED + kk + k]);
    }
    __syncthreads();

    v16bf a;
    const unsigned short* arow = &As[(wave * 16 + m16) * 32];
#pragma unroll
    for (int e = 0; e < 16; ++e) a[e] = bf16_bits(arow[a_frag_k(e, lane)]);
#pragma unroll
    for (int t = 0; t < 4; ++t) {
      v16bf b;
      const unsigned short* brow = &Bs[(t * 16 + m16) * 32];
#pragma unroll
      for (int e = 0; e < 16; ++e) b[e] = bf16_bits(brow[a_frag_k(e, lane)]);
      acc[t] = wmma_bf16(a, b, acc[t]);
    }
  }

  const int moff = (lane & 16) ? 8 : 0;
#pragma unroll
  for (int t = 0; t < 4; ++t) {
    const int f = Nbase + t * 16 + m16;
    const float bv = bias[f];
#pragma unroll
    for (int i = 0; i < 8; ++i) {
      const int R  = Mbase + wave * 16 + i + moff;
      const int nn = R >> 4, bb = R & 15;
      const float val = acc[t][i] + bv;
      out[(size_t)R * EMBED + f] = mask[bb * N_NODE + nn] ? val : 0.0f;
    }
  }
}

// ---------------------------------------------------------------------------
extern "C" void kernel_launch(void* const* d_in, const int* in_sizes, int n_in,
                              void* d_out, int out_size, void* d_ws,
                              size_t ws_size, hipStream_t stream) {
  const float*         query     = (const float*)d_in[0];
  const float*         attn_bias = (const float*)d_in[1];
  const unsigned char* mask      = (const unsigned char*)d_in[2];
  const float*         in_w      = (const float*)d_in[3];
  const float*         in_b      = (const float*)d_in[4];
  const float*         out_w     = (const float*)d_in[5];
  const float*         out_b     = (const float*)d_in[6];
  float*               out       = (float*)d_out;

  const size_t tensor_elems = (size_t)GTOT * N_NODE * HDIM;  // 4M bf16 = 8MB
  unsigned short* qws = (unsigned short*)d_ws;
  unsigned short* kws = qws + tensor_elems;
  unsigned short* vws = kws + tensor_elems;
  unsigned short* aws = vws + tensor_elems;                   // 32MB total

  // 1) QKV projection: M=8192, N=1536, K=512 (tile 128x64)
  qkv_gemm_kernel<<<dim3(NROWS / 128, 1536 / 64), 256, 0, stream>>>(
      query, in_w, in_b, qws, kws, vws);

  // 2) attention: 512 heads x 4 row-strips of 128; 292KB dynamic LDS
  const size_t shmem = (size_t)LDS_SS_OFF + (size_t)128 * N_NODE * 4;
  attn_kernel<<<dim3(GTOT, N_NODE / 128), 256, shmem, stream>>>(
      qws, kws, vws, attn_bias, aws);

  // 3) output projection: M=8192, N=512, K=512 (tile 128x64)
  out_proj_kernel<<<dim3(NROWS / 128, EMBED / 64), 256, 0, stream>>>(
      aws, out_w, out_b, mask, out);
}
